// WindowMSA_48747878810063
// MI455X (gfx1250) — matrix operand order repaired and checked by reading, not compiled, not run on hardware
//
#include <hip/hip_runtime.h>

typedef __bf16 bf16;
typedef __attribute__((ext_vector_type(8)))  bf16  bf16x8;
typedef __attribute__((ext_vector_type(16))) bf16  bf16x16;
typedef __attribute__((ext_vector_type(8)))  float f32x8;

// ---------------- workspace layout (bytes) ----------------
#define WS_QKVW   0            // 256*128 bf16 = 65536
#define WS_SKIPW  65536        // 128*128 bf16 = 32768 (scale folded in)
#define WS_PROJW  98304        // 128*128 bf16 = 32768
#define WS_QB     131072       // 256 f32
#define WS_SB     132096       // 128 f32 (scale folded in)
#define WS_PB     132608       // 128 f32
#define WS_RELB   133120       // 4*49*49 f32 = 38416
// total ~171.5 KB

#define LDX   136   // bf16 stride for 64x128 row-major tiles (+8 pad)
#define LDV   72    // bf16 stride for V^T (128 x 64)
#define LDP   72    // bf16 stride for probs (64 x 64)
#define LDS_S 68    // f32 stride for scores (64 x 64)

static __device__ inline f32x8 wmma_bf16(bf16x16 a, bf16x16 b, f32x8 c) {
  return __builtin_amdgcn_wmma_f32_16x16x32_bf16(false, a, false, b, (short)0, c,
                                                 false, false);
}

// A-fragment (16x32, bf16): lane<16 -> M=lane, K in {0..7,16..23};
// lane>=16 -> M=lane-16, K in {8..15,24..31}.  (ISA 7.12.2)
static __device__ inline bf16x16 frag_a(const bf16* base, int m0, int k0, int ld, int lane) {
  const int half = lane >> 4, rr = lane & 15;
  const bf16* p = base + (m0 + rr) * ld + k0;
  bf16x8 lo = *(const bf16x8*)(p + half * 8);
  bf16x8 hi = *(const bf16x8*)(p + 16 + half * 8);
  return __builtin_shufflevector(lo, hi, 0,1,2,3,4,5,6,7,8,9,10,11,12,13,14,15);
}

// B-fragment (32x16, bf16), loaded from Bt = B^T stored row-major (16 x 32):
// lane L holds column N=L%16, K in {8h..8h+7, 16+8h..16+8h+7}, h=L/16.
static __device__ inline bf16x16 frag_b(const bf16* base, int n0, int k0, int ld, int lane) {
  const int half = lane >> 4, rr = lane & 15;
  const bf16* p = base + (n0 + rr) * ld + k0;
  bf16x8 lo = *(const bf16x8*)(p + half * 8);
  bf16x8 hi = *(const bf16x8*)(p + 16 + half * 8);
  return __builtin_shufflevector(lo, hi, 0,1,2,3,4,5,6,7,8,9,10,11,12,13,14,15);
}

// ---------------- prep kernel: bf16 weights, biases, rel-pos bias table ----
__global__ void wmsa_prep(const float* __restrict__ qkv_w, const float* __restrict__ qkv_b,
                          const float* __restrict__ skip_w, const float* __restrict__ skip_b,
                          const float* __restrict__ bias_table,
                          const float* __restrict__ proj_w, const float* __restrict__ proj_b,
                          unsigned char* __restrict__ ws) {
  bf16*  qkvw = (bf16*)(ws + WS_QKVW);
  bf16*  skpw = (bf16*)(ws + WS_SKIPW);
  bf16*  prjw = (bf16*)(ws + WS_PROJW);
  float* qb   = (float*)(ws + WS_QB);
  float* sb   = (float*)(ws + WS_SB);
  float* pb   = (float*)(ws + WS_PB);
  float* relb = (float*)(ws + WS_RELB);

  const float scale = 0.17677669529663687f;  // 32^-0.5 folded into q path
  const int tid = blockIdx.x * blockDim.x + threadIdx.x;
  const int nth = gridDim.x * blockDim.x;

  for (int i = tid; i < 256 * 128; i += nth) qkvw[i] = (bf16)qkv_w[i];
  for (int i = tid; i < 128 * 128; i += nth) skpw[i] = (bf16)(skip_w[i] * scale);
  for (int i = tid; i < 128 * 128; i += nth) prjw[i] = (bf16)proj_w[i];
  for (int i = tid; i < 256; i += nth) qb[i] = qkv_b[i];
  for (int i = tid; i < 128; i += nth) sb[i] = skip_b[i] * scale;
  for (int i = tid; i < 128; i += nth) pb[i] = proj_b[i];

  // rel_idx[n][m] = c[n] + c[48-m], c[p] = 13*(p/7) + (p%7); relb[h][n][m]
  for (int i = tid; i < 4 * 49 * 49; i += nth) {
    int h = i / (49 * 49);
    int rem = i % (49 * 49);
    int n = rem / 49, m = rem % 49;
    int cn = 13 * (n / 7) + (n % 7);
    int fm = 48 - m;
    int cm = 13 * (fm / 7) + (fm % 7);
    relb[i] = bias_table[(cn + cm) * 4 + h];
  }
}

// ---------------- fused window attention kernel ---------------------------
__global__ __launch_bounds__(256) void wmsa_main(const float* __restrict__ x,
                                                 const float* __restrict__ skip_x,
                                                 const float* __restrict__ mask,
                                                 const unsigned char* __restrict__ ws,
                                                 float* __restrict__ out) {
  // carved LDS: regions overlapped where lifetimes are disjoint
  __shared__ __align__(16) unsigned char smem[105472];
  bf16*  Xs  = (bf16*)(smem + 0);       // x bf16, 64 x LDX   (dead after GEMM1)
  bf16*  Os  = (bf16*)(smem + 0);       // head outputs, 64 x LDX (aliases Xs)
  bf16*  SKs = (bf16*)(smem + 17408);   // skip_x bf16 (dead after GEMM2)
  bf16*  Ps  = (bf16*)(smem + 17408);   // probs bf16, 64 x LDP (aliases SKs)
  bf16*  Ks  = (bf16*)(smem + 34816);   // k, 64 x LDX (rows=token, cols=feat)
  bf16*  VTs = (bf16*)(smem + 52224);   // v^T, 128 x LDV (rows=h*32+d, cols=m)
  bf16*  Qs  = (bf16*)(smem + 70656);   // q, 64 x LDX
  float* Ss  = (float*)(smem + 88064);  // scores f32, 64 x LDS_S

  const int b    = blockIdx.x;
  const int tid  = threadIdx.x;
  const int lane = tid & 31;
  const int wv   = tid >> 5;        // wave id 0..7
  const int half = lane >> 4;
  const int r    = lane & 15;

  const bf16*  qkvw = (const bf16*)(ws + WS_QKVW);
  const bf16*  skpw = (const bf16*)(ws + WS_SKIPW);
  const bf16*  prjw = (const bf16*)(ws + WS_PROJW);
  const float* qb   = (const float*)(ws + WS_QB);
  const float* sb   = (const float*)(ws + WS_SB);
  const float* pb   = (const float*)(ws + WS_PB);
  const float* relb = (const float*)(ws + WS_RELB);

  // ---- stage x / skip_x as bf16 (rows 49..63 zero) ----
  const float4* xg = (const float4*)(x + (size_t)b * 49 * 128);
  const float4* sg = (const float4*)(skip_x + (size_t)b * 49 * 128);
  for (int e = tid; e < 64 * 32; e += 256) {
    int row = e >> 5, c4 = e & 31;
    float4 vx = make_float4(0.f, 0.f, 0.f, 0.f);
    float4 vs = make_float4(0.f, 0.f, 0.f, 0.f);
    if (row < 49) { vx = xg[row * 32 + c4]; vs = sg[row * 32 + c4]; }
    int o = row * LDX + c4 * 4;
    Xs[o] = (bf16)vx.x; Xs[o + 1] = (bf16)vx.y; Xs[o + 2] = (bf16)vx.z; Xs[o + 3] = (bf16)vx.w;
    SKs[o] = (bf16)vs.x; SKs[o + 1] = (bf16)vs.y; SKs[o + 2] = (bf16)vs.z; SKs[o + 3] = (bf16)vs.w;
  }
  __syncthreads();

  // ---- GEMM1: kv = x @ qkv_w^T  (49x128 @ 128x256), epilogue scatters k / v^T
  for (int ntl = 0; ntl < 2; ++ntl) {
    const int nt = wv * 2 + ntl;  // 0..15 (output cols 16*nt)
    f32x8 acc[4];
#pragma unroll
    for (int mt = 0; mt < 4; ++mt) acc[mt] = (f32x8){0.f,0.f,0.f,0.f,0.f,0.f,0.f,0.f};
#pragma unroll
    for (int k = 0; k < 4; ++k) {
      bf16x16 bf = frag_b(qkvw, nt * 16, k * 32, 128, lane);
#pragma unroll
      for (int mt = 0; mt < 4; ++mt) {
        bf16x16 af = frag_a(Xs, mt * 16, k * 32, LDX, lane);
        acc[mt] = wmma_bf16(af, bf, acc[mt]);
      }
    }
    const int col = nt * 16 + r;
    const float bias = qb[col];
#pragma unroll
    for (int mt = 0; mt < 4; ++mt) {
#pragma unroll
      for (int i = 0; i < 8; ++i) {
        int row = mt * 16 + i + half * 8;
        float v = acc[mt][i] + bias;
        if (col < 128) {
          Ks[row * LDX + col] = (bf16)v;                 // k, row-major
        } else {
          int c = col - 128, hh = c >> 5, d = c & 31;
          VTs[(hh * 32 + d) * LDV + row] = (bf16)v;      // v^T per head
        }
      }
    }
  }

  // ---- GEMM2: q = skip_x @ skip_w^T (scale pre-folded) ----
  {
    const int nt = wv;  // 0..7
    f32x8 acc[4];
#pragma unroll
    for (int mt = 0; mt < 4; ++mt) acc[mt] = (f32x8){0.f,0.f,0.f,0.f,0.f,0.f,0.f,0.f};
#pragma unroll
    for (int k = 0; k < 4; ++k) {
      bf16x16 bf = frag_b(skpw, nt * 16, k * 32, 128, lane);
#pragma unroll
      for (int mt = 0; mt < 4; ++mt) {
        bf16x16 af = frag_a(SKs, mt * 16, k * 32, LDX, lane);
        acc[mt] = wmma_bf16(af, bf, acc[mt]);
      }
    }
    const int col = nt * 16 + r;
    const float bias = sb[col];
#pragma unroll
    for (int mt = 0; mt < 4; ++mt)
#pragma unroll
      for (int i = 0; i < 8; ++i)
        Qs[(mt * 16 + i + half * 8) * LDX + col] = (bf16)(acc[mt][i] + bias);
  }
  __syncthreads();

  const float* maskb = mask + (size_t)(b & 63) * 49 * 49;

  // softmax mapping: 4 lanes per row -> all 256 threads active
  const int sm_n = tid >> 2;   // row 0..63
  const int sm_j = tid & 3;    // sub-lane 0..3 (consecutive lanes of one wave)

  // ---- per-head attention ----
  for (int h = 0; h < 4; ++h) {
    // S = q_h @ k_h^T : K = hd = 32 -> exactly one WMMA per 16x16 tile
#pragma unroll
    for (int tl = 0; tl < 2; ++tl) {
      const int t = wv * 2 + tl, mt = t >> 2, nt = t & 3;
      bf16x16 af = frag_a(Qs, mt * 16, h * 32, LDX, lane);
      bf16x16 bf = frag_b(Ks, nt * 16, h * 32, LDX, lane);  // Bt = rows of K
      f32x8 acc = (f32x8){0.f,0.f,0.f,0.f,0.f,0.f,0.f,0.f};
      acc = wmma_bf16(af, bf, acc);
#pragma unroll
      for (int i = 0; i < 8; ++i)
        Ss[(mt * 16 + i + half * 8) * LDS_S + nt * 16 + r] = acc[i];
    }
    __syncthreads();

    // softmax over m (+rel bias, +mask); 4 lanes cooperate per row via
    // wave32 shuffles; 13 register-resident elements per lane, pad cols -> 0
    if (sm_n < 49) {
      const float* rb = relb + ((size_t)h * 49 + sm_n) * 49;
      const float* mk = maskb + sm_n * 49;
      float sv[13];
      float mx = -1e30f;
#pragma unroll
      for (int t = 0; t < 13; ++t) {
        int m = sm_j + 4 * t;
        float s = -1e30f;
        if (m < 49) s = Ss[sm_n * LDS_S + m] + rb[m] + mk[m];
        sv[t] = s;
        mx = fmaxf(mx, s);
      }
      mx = fmaxf(mx, __shfl_xor(mx, 1, 32));
      mx = fmaxf(mx, __shfl_xor(mx, 2, 32));
      float sum = 0.f;
#pragma unroll
      for (int t = 0; t < 13; ++t) {
        int m = sm_j + 4 * t;
        float e = (m < 49) ? __expf(sv[t] - mx) : 0.f;
        sv[t] = e;
        sum += e;
      }
      sum += __shfl_xor(sum, 1, 32);
      sum += __shfl_xor(sum, 2, 32);
      const float inv = 1.f / sum;
#pragma unroll
      for (int t = 0; t < 13; ++t) {
        int m = sm_j + 4 * t;
        if (m < 64) Ps[sm_n * LDP + m] = (bf16)(sv[t] * inv);
      }
      // zero the pad columns 49..63 (each lane covers its stride-4 subset)
      for (int m = 49 + ((sm_j - 49) & 3); m < 64; m += 4) Ps[sm_n * LDP + m] = (bf16)0.f;
    } else {
      for (int m = sm_j; m < 64; m += 4) Ps[sm_n * LDP + m] = (bf16)0.f;
    }
    __syncthreads();

    // O_h = P_h @ v_h : K = 64 (padded) -> 2 WMMAs; Bt = v^T rows
    {
      const int mt = wv >> 1, nt = wv & 1;
      f32x8 acc = (f32x8){0.f,0.f,0.f,0.f,0.f,0.f,0.f,0.f};
#pragma unroll
      for (int ks = 0; ks < 2; ++ks) {
        bf16x16 af = frag_a(Ps, mt * 16, ks * 32, LDP, lane);
        bf16x16 bf = frag_b(VTs + h * 32 * LDV, nt * 16, ks * 32, LDV, lane);
        acc = wmma_bf16(af, bf, acc);
      }
      const int col = h * 32 + nt * 16 + r;
#pragma unroll
      for (int i = 0; i < 8; ++i)
        Os[(mt * 16 + i + half * 8) * LDX + col] = (bf16)acc[i];
    }
    __syncthreads();
  }

  // ---- GEMM5: out = O @ proj_w^T + proj_b ----
  {
    const int nt = wv;
    f32x8 acc[4];
#pragma unroll
    for (int mt = 0; mt < 4; ++mt) acc[mt] = (f32x8){0.f,0.f,0.f,0.f,0.f,0.f,0.f,0.f};
#pragma unroll
    for (int k = 0; k < 4; ++k) {
      bf16x16 bf = frag_b(prjw, nt * 16, k * 32, 128, lane);
#pragma unroll
      for (int mt = 0; mt < 4; ++mt) {
        bf16x16 af = frag_a(Os, mt * 16, k * 32, LDX, lane);
        acc[mt] = wmma_bf16(af, bf, acc[mt]);
      }
    }
    const int col = nt * 16 + r;
    const float bias = pb[col];
    float* og = out + (size_t)b * 49 * 128;
#pragma unroll
    for (int mt = 0; mt < 4; ++mt)
#pragma unroll
      for (int i = 0; i < 8; ++i) {
        int row = mt * 16 + i + half * 8;
        if (row < 49) og[row * 128 + col] = acc[mt][i] + bias;
      }
  }
}

extern "C" void kernel_launch(void* const* d_in, const int* in_sizes, int n_in,
                              void* d_out, int out_size, void* d_ws, size_t ws_size,
                              hipStream_t stream) {
  const float* x          = (const float*)d_in[0];
  const float* skip_x     = (const float*)d_in[1];
  const float* mask       = (const float*)d_in[2];
  const float* qkv_w      = (const float*)d_in[3];
  const float* qkv_b      = (const float*)d_in[4];
  const float* skip_w     = (const float*)d_in[5];
  const float* skip_b     = (const float*)d_in[6];
  const float* bias_table = (const float*)d_in[7];
  const float* proj_w     = (const float*)d_in[8];
  const float* proj_b     = (const float*)d_in[9];

  wmsa_prep<<<64, 256, 0, stream>>>(qkv_w, qkv_b, skip_w, skip_b, bias_table,
                                    proj_w, proj_b, (unsigned char*)d_ws);
  wmsa_main<<<4096, 256, 0, stream>>>(x, skip_x, mask, (const unsigned char*)d_ws,
                                      (float*)d_out);
}
